// Actor_61469571940534
// MI455X (gfx1250) — matrix operand order repaired and checked by reading, not compiled, not run on hardware
//
#include <hip/hip_runtime.h>
#include <math.h>

// ---------------------------------------------------------------------------
// MoE actor: blend-gated 2-expert MLP. Fused WMMA kernel, 32-row batch tile
// per block (two 16-row m-subtiles per wave) so each B (weight) fragment
// fetched from L2 is reused by two v_wmma issues -> halves L2 weight traffic.
// ---------------------------------------------------------------------------

typedef __attribute__((ext_vector_type(16))) _Float16 v16h;
typedef __attribute__((ext_vector_type(8)))  _Float16 v8h;
typedef __attribute__((ext_vector_type(8)))  float    v8f;

#define NUM_INPUTS 376
#define K_IN_PAD   384      // 376 padded to multiple of 32
#define NUM_ACT    17
#define ACT_PAD    32
#define L1D        256
#define L2D        128
#define BHIDD      128
#define NEXP       2
#define BATCH      65536
#define MTILE      32       // rows per block (2 m-subtiles of 16)

// LDS row strides (in halves); multiples of 8 -> 16B-aligned b128 loads
#define S_STRIDE   392      // states tile 32 x 384 (+8 pad vs bank conflicts)
#define A_STRIDE   264      // G1 (32x128) / H1 (32x256)
#define B_STRIDE   136      // G2 / H2 (32x128)

// Workspace layout (offsets in _Float16 elements); all 32B aligned in bytes.
#define OFF_BW1 0                       // 128 x 384
#define OFF_BW2 (OFF_BW1 + 128*384)     // 128 x 128
#define OFF_BWO (OFF_BW2 + 128*128)     // 16  x 128
#define OFF_EW1 (OFF_BWO + 16*128)      // 2 x 256 x 384
#define OFF_EW2 (OFF_EW1 + 2*256*384)   // 2 x 128 x 256
#define OFF_EWM (OFF_EW2 + 2*128*256)   // 2 x 32  x 128

// ---------------------------------------------------------------------------
// f32 -> f16 weight conversion with O/K padding (pads zero-filled every call)
// src layout: [E, O, K] row-major f32 ; dst: [E, Opad, Kpad] row-major f16
// ---------------------------------------------------------------------------
__global__ void convert_pad_kernel(const float* __restrict__ src,
                                   _Float16* __restrict__ dst,
                                   int E, int O, int K, int Opad, int Kpad) {
  int total = E * Opad * Kpad;
  for (int idx = blockIdx.x * blockDim.x + threadIdx.x; idx < total;
       idx += gridDim.x * blockDim.x) {
    int e   = idx / (Opad * Kpad);
    int rem = idx - e * (Opad * Kpad);
    int o   = rem / Kpad;
    int k   = rem - o * Kpad;
    float v = (o < O && k < K) ? src[((size_t)e * O + o) * K + k] : 0.0f;
    dst[idx] = (_Float16)v;
  }
}

// ---------------------------------------------------------------------------
// Fragment loaders (wave32, contiguous k order shared by A and B)
// ---------------------------------------------------------------------------
// A 16x32 f16: lane row = lane&15 ; kb = 16*(lane>>4):
//   halves 0..7  -> K = kt*32 + kb .. +7      (one ds_load_b128)
//   halves 8..15 -> K = kt*32 + kb + 8 .. 15  (one ds_load_b128, +16B)
__device__ __forceinline__ v16h frag_a_lds(const _Float16* base, int stride,
                                           int kt, int lane) {
  int row = lane & 15;
  int kb  = (kt << 5) + ((lane >> 4) << 4);
  const _Float16* p = base + row * stride + kb;
  v8h lo = *(const v8h*)p;
  v8h hi = *(const v8h*)(p + 8);
  return __builtin_shufflevector(lo, hi, 0, 1, 2, 3, 4, 5, 6, 7,
                                 8, 9, 10, 11, 12, 13, 14, 15);
}

// B 32x16 f16: lane column = lane&15 ; lanes 0-15 hold K 0..15 of the tile,
// lanes 16-31 hold K 16..31, contiguous -> one 32B load from the row-major
// f16 weight W[Opad, Kpad] (column n of B == row n of W).
__device__ __forceinline__ v16h frag_b_gmem(const _Float16* w, int kpad,
                                            int n0, int kt, int lane) {
  int col = n0 + (lane & 15);
  int kb  = (kt << 5) + ((lane >> 4) << 4);
  return *(const v16h*)(w + (size_t)col * kpad + kb);
}

__device__ __forceinline__ v8f wmma_f16(v16h a, v16h b, v8f c) {
  return __builtin_amdgcn_wmma_f32_16x16x32_f16(false, a, false, b,
                                                (short)0, c, false, false);
}

// ---------------------------------------------------------------------------
// Fused MoE actor kernel: 32-row batch tile per 128-thread block (4 waves).
// Waves split the N dimension; each wave runs both 16-row m-subtiles so every
// B fragment is reused by 2 WMMAs.
// ---------------------------------------------------------------------------
__global__ __launch_bounds__(128) void Actor_61469571940534_kernel(
    const float* __restrict__ states,
    const float* __restrict__ bb1, const float* __restrict__ bb2,
    const float* __restrict__ bbo,
    const float* __restrict__ eb1, const float* __restrict__ eb2,
    const float* __restrict__ ebm,
    const _Float16* __restrict__ bw1h, const _Float16* __restrict__ bw2h,
    const _Float16* __restrict__ bwoh, const _Float16* __restrict__ ew1h,
    const _Float16* __restrict__ ew2h, const _Float16* __restrict__ ewmh,
    float* __restrict__ out) {
  __shared__ _Float16 S[MTILE * S_STRIDE];      // states tile, f16, K padded
  __shared__ _Float16 BufA[MTILE * A_STRIDE];   // G1 then H1
  __shared__ _Float16 BufB[MTILE * B_STRIDE];   // G2 then H2
  __shared__ float    Cf[MTILE * 2];            // logits -> blend coeffs

  const int tid  = threadIdx.x;
  const int lane = tid & 31;
  const int wv   = tid >> 5;
  const int colr = lane & 15;
  const int hi   = lane >> 4;
  const int row0 = blockIdx.x * MTILE;

  // ---- Stage 0: stage 32x376 f32 states into LDS as f16, zero-pad to 384 --
  for (int idx = tid; idx < MTILE * 96; idx += 128) {
    int r  = idx / 96;
    int c4 = idx - r * 96;                 // float4 group
    _Float16* dst = &S[r * S_STRIDE + (c4 << 2)];
    if (c4 < 94) {                         // 94*4 == 376, 16B-aligned loads
      float4 v = *(const float4*)(states + (size_t)(row0 + r) * NUM_INPUTS +
                                  (c4 << 2));
      dst[0] = (_Float16)v.x; dst[1] = (_Float16)v.y;
      dst[2] = (_Float16)v.z; dst[3] = (_Float16)v.w;
    } else {                               // cols 376..383 -> 0
      dst[0] = (_Float16)0.f; dst[1] = (_Float16)0.f;
      dst[2] = (_Float16)0.f; dst[3] = (_Float16)0.f;
    }
  }
  __syncthreads();

  // ---- Stage 1: G1 = relu(S @ bw1^T + bb1)  [32 x 128], 8 ntiles / 4 waves
  for (int t = 0; t < 2; ++t) {
    int n0 = (wv * 2 + t) * 16;
    v8f acc0 = {}, acc1 = {};
    for (int kt = 0; kt < 12; ++kt) {
      v16h b  = frag_b_gmem(bw1h, K_IN_PAD, n0, kt, lane);
      v16h a0 = frag_a_lds(S, S_STRIDE, kt, lane);
      v16h a1 = frag_a_lds(S + 16 * S_STRIDE, S_STRIDE, kt, lane);
      acc0 = wmma_f16(a0, b, acc0);
      acc1 = wmma_f16(a1, b, acc1);
    }
    float bias = bb1[n0 + colr];
#pragma unroll
    for (int r = 0; r < 8; ++r) {
      float v0 = acc0[r] + bias, v1 = acc1[r] + bias;
      BufA[(r + 8 * hi) * A_STRIDE + n0 + colr] =
          (_Float16)(v0 > 0.f ? v0 : 0.f);
      BufA[(16 + r + 8 * hi) * A_STRIDE + n0 + colr] =
          (_Float16)(v1 > 0.f ? v1 : 0.f);
    }
  }
  __syncthreads();

  // ---- Stage 2: G2 = relu(G1 @ bw2^T + bb2)  [32 x 128] --------------------
  for (int t = 0; t < 2; ++t) {
    int n0 = (wv * 2 + t) * 16;
    v8f acc0 = {}, acc1 = {};
    for (int kt = 0; kt < 4; ++kt) {
      v16h b  = frag_b_gmem(bw2h, BHIDD, n0, kt, lane);
      v16h a0 = frag_a_lds(BufA, A_STRIDE, kt, lane);
      v16h a1 = frag_a_lds(BufA + 16 * A_STRIDE, A_STRIDE, kt, lane);
      acc0 = wmma_f16(a0, b, acc0);
      acc1 = wmma_f16(a1, b, acc1);
    }
    float bias = bb2[n0 + colr];
#pragma unroll
    for (int r = 0; r < 8; ++r) {
      float v0 = acc0[r] + bias, v1 = acc1[r] + bias;
      BufB[(r + 8 * hi) * B_STRIDE + n0 + colr] =
          (_Float16)(v0 > 0.f ? v0 : 0.f);
      BufB[(16 + r + 8 * hi) * B_STRIDE + n0 + colr] =
          (_Float16)(v1 > 0.f ? v1 : 0.f);
    }
  }
  __syncthreads();

  // ---- Stage 3: expert logits (N=16 tile, 2 valid) -> softmax coeffs -------
  if (wv < 2) {                      // wave wv handles m-subtile wv
    const _Float16* src = BufB + wv * 16 * B_STRIDE;
    v8f acc = {};
    for (int kt = 0; kt < 4; ++kt) {
      v16h a = frag_a_lds(src, B_STRIDE, kt, lane);
      v16h b = frag_b_gmem(bwoh, BHIDD, 0, kt, lane);
      acc = wmma_f16(a, b, acc);
    }
    if (colr < NEXP) {
#pragma unroll
      for (int r = 0; r < 8; ++r)
        Cf[(wv * 16 + r + 8 * hi) * 2 + colr] = acc[r] + bbo[colr];
    }
  }
  __syncthreads();
  if (tid < MTILE) {
    float l0 = Cf[tid * 2], l1 = Cf[tid * 2 + 1];
    float mx = fmaxf(l0, l1);
    float e0 = __expf(l0 - mx), e1 = __expf(l1 - mx);
    float inv = 1.f / (e0 + e1);
    Cf[tid * 2]     = e0 * inv;
    Cf[tid * 2 + 1] = e1 * inv;
  }
  __syncthreads();

  // ---- Stage 4: H1 = relu(sum_e c_e (S @ ew1_e^T + eb1_e))  [32 x 256] -----
  for (int t = 0; t < 4; ++t) {
    int n0 = (wv * 4 + t) * 16;
    v8f e0m0 = {}, e0m1 = {}, e1m0 = {}, e1m1 = {};
    for (int kt = 0; kt < 12; ++kt) {
      v16h b0 = frag_b_gmem(ew1h,                          K_IN_PAD, n0, kt, lane);
      v16h b1 = frag_b_gmem(ew1h + (size_t)L1D * K_IN_PAD, K_IN_PAD, n0, kt, lane);
      v16h a0 = frag_a_lds(S, S_STRIDE, kt, lane);
      v16h a1 = frag_a_lds(S + 16 * S_STRIDE, S_STRIDE, kt, lane);
      e0m0 = wmma_f16(a0, b0, e0m0);
      e0m1 = wmma_f16(a1, b0, e0m1);
      e1m0 = wmma_f16(a0, b1, e1m0);
      e1m1 = wmma_f16(a1, b1, e1m1);
    }
    float bias0 = eb1[n0 + colr];
    float bias1 = eb1[L1D + n0 + colr];
#pragma unroll
    for (int r = 0; r < 8; ++r) {
      int m0 = r + 8 * hi, m1 = 16 + r + 8 * hi;
      float2 cA = *(const float2*)&Cf[m0 * 2];
      float2 cB = *(const float2*)&Cf[m1 * 2];
      float v0 = cA.x * (e0m0[r] + bias0) + cA.y * (e1m0[r] + bias1);
      float v1 = cB.x * (e0m1[r] + bias0) + cB.y * (e1m1[r] + bias1);
      BufA[m0 * A_STRIDE + n0 + colr] = (_Float16)(v0 > 0.f ? v0 : 0.f);
      BufA[m1 * A_STRIDE + n0 + colr] = (_Float16)(v1 > 0.f ? v1 : 0.f);
    }
  }
  __syncthreads();

  // ---- Stage 5: H2 = relu(sum_e c_e (H1 @ ew2_e^T + eb2_e))  [32 x 128] ----
  for (int t = 0; t < 2; ++t) {
    int n0 = (wv * 2 + t) * 16;
    v8f e0m0 = {}, e0m1 = {}, e1m0 = {}, e1m1 = {};
    for (int kt = 0; kt < 8; ++kt) {
      v16h b0 = frag_b_gmem(ew2h,                     L1D, n0, kt, lane);
      v16h b1 = frag_b_gmem(ew2h + (size_t)L2D * L1D, L1D, n0, kt, lane);
      v16h a0 = frag_a_lds(BufA, A_STRIDE, kt, lane);
      v16h a1 = frag_a_lds(BufA + 16 * A_STRIDE, A_STRIDE, kt, lane);
      e0m0 = wmma_f16(a0, b0, e0m0);
      e0m1 = wmma_f16(a1, b0, e0m1);
      e1m0 = wmma_f16(a0, b1, e1m0);
      e1m1 = wmma_f16(a1, b1, e1m1);
    }
    float bias0 = eb2[n0 + colr];
    float bias1 = eb2[L2D + n0 + colr];
#pragma unroll
    for (int r = 0; r < 8; ++r) {
      int m0 = r + 8 * hi, m1 = 16 + r + 8 * hi;
      float2 cA = *(const float2*)&Cf[m0 * 2];
      float2 cB = *(const float2*)&Cf[m1 * 2];
      float v0 = cA.x * (e0m0[r] + bias0) + cA.y * (e1m0[r] + bias1);
      float v1 = cB.x * (e0m1[r] + bias0) + cB.y * (e1m1[r] + bias1);
      BufB[m0 * B_STRIDE + n0 + colr] = (_Float16)(v0 > 0.f ? v0 : 0.f);
      BufB[m1 * B_STRIDE + n0 + colr] = (_Float16)(v1 > 0.f ? v1 : 0.f);
    }
  }
  __syncthreads();

  // ---- Stage 6: mu = tanh(sum_e c_e (H2 @ ewm_e^T + ebm_e))  [32 x 17] -----
  {
    int nt = wv & 1;                 // n-tile 0..1 (cols 0..15 / 16..31)
    int ms = wv >> 1;                // m-subtile 0..1
    const _Float16* src = BufB + ms * 16 * B_STRIDE;
    int n0 = nt * 16;
    v8f a0 = {}, a1 = {};
    for (int kt = 0; kt < 4; ++kt) {
      v16h a  = frag_a_lds(src, B_STRIDE, kt, lane);
      v16h b0 = frag_b_gmem(ewmh,                         L2D, n0, kt, lane);
      v16h b1 = frag_b_gmem(ewmh + (size_t)ACT_PAD * L2D, L2D, n0, kt, lane);
      a0 = wmma_f16(a, b0, a0);
      a1 = wmma_f16(a, b1, a1);
    }
    int n = n0 + colr;
    if (n < NUM_ACT) {
      float bias0 = ebm[n];
      float bias1 = ebm[NUM_ACT + n];
#pragma unroll
      for (int r = 0; r < 8; ++r) {
        int m = ms * 16 + r + 8 * hi;
        float2 c = *(const float2*)&Cf[m * 2];
        float v = c.x * (a0[r] + bias0) + c.y * (a1[r] + bias1);
        out[(size_t)(row0 + m) * NUM_ACT + n] = tanhf(v);
      }
    }
  }
}

// ---------------------------------------------------------------------------
// Launch: convert weights to padded f16 in d_ws, then run the fused kernel.
// Input order: states, bw1, bb1, bw2, bb2, bwo, bbo, ew1, eb1, ew2, eb2,
//              ewm, ebm
// ---------------------------------------------------------------------------
extern "C" void kernel_launch(void* const* d_in, const int* in_sizes, int n_in,
                              void* d_out, int out_size, void* d_ws,
                              size_t ws_size, hipStream_t stream) {
  (void)in_sizes; (void)n_in; (void)out_size; (void)ws_size;

  const float* states = (const float*)d_in[0];
  const float* bw1 = (const float*)d_in[1];
  const float* bb1 = (const float*)d_in[2];
  const float* bw2 = (const float*)d_in[3];
  const float* bb2 = (const float*)d_in[4];
  const float* bwo = (const float*)d_in[5];
  const float* bbo = (const float*)d_in[6];
  const float* ew1 = (const float*)d_in[7];
  const float* eb1 = (const float*)d_in[8];
  const float* ew2 = (const float*)d_in[9];
  const float* eb2 = (const float*)d_in[10];
  const float* ewm = (const float*)d_in[11];
  const float* ebm = (const float*)d_in[12];
  float* out = (float*)d_out;

  _Float16* wsh  = (_Float16*)d_ws;
  _Float16* bw1h = wsh + OFF_BW1;
  _Float16* bw2h = wsh + OFF_BW2;
  _Float16* bwoh = wsh + OFF_BWO;
  _Float16* ew1h = wsh + OFF_EW1;
  _Float16* ew2h = wsh + OFF_EW2;
  _Float16* ewmh = wsh + OFF_EWM;

  auto cgrid = [](int total) { return (total + 255) / 256; };

  convert_pad_kernel<<<cgrid(1 * 128 * 384), 256, 0, stream>>>(
      bw1, bw1h, 1, BHIDD, NUM_INPUTS, BHIDD, K_IN_PAD);
  convert_pad_kernel<<<cgrid(1 * 128 * 128), 256, 0, stream>>>(
      bw2, bw2h, 1, BHIDD, BHIDD, BHIDD, BHIDD);
  convert_pad_kernel<<<cgrid(1 * 16 * 128), 256, 0, stream>>>(
      bwo, bwoh, 1, NEXP, BHIDD, 16, BHIDD);
  convert_pad_kernel<<<cgrid(2 * 256 * 384), 256, 0, stream>>>(
      ew1, ew1h, NEXP, L1D, NUM_INPUTS, L1D, K_IN_PAD);
  convert_pad_kernel<<<cgrid(2 * 128 * 256), 256, 0, stream>>>(
      ew2, ew2h, NEXP, L2D, L1D, L2D, L1D);
  convert_pad_kernel<<<cgrid(2 * 32 * 128), 256, 0, stream>>>(
      ewm, ewmh, NEXP, NUM_ACT, L2D, ACT_PAD, L2D);

  Actor_61469571940534_kernel<<<BATCH / MTILE, 128, 0, stream>>>(
      states, bb1, bb2, bbo, eb1, eb2, ebm,
      bw1h, bw2h, bwoh, ew1h, ew2h, ewmh, out);
}
